// SelfAttention_609885356591
// MI455X (gfx1250) — matrix operand order repaired and checked by reading, not compiled, not run on hardware
//
#include <hip/hip_runtime.h>
#include <math.h>

typedef unsigned short u16;
typedef __attribute__((ext_vector_type(16))) __bf16 v16bf;
typedef __attribute__((ext_vector_type(8)))  float  v8f;

union ABFrag { uint4 u4[2]; v16bf v; };

__device__ __forceinline__ u16 f2bf(float f) {
  union { float f; unsigned u; } c; c.f = f;
  unsigned r = c.u + 0x7FFFu + ((c.u >> 16) & 1u);   // round-to-nearest-even
  return (u16)(r >> 16);
}

// B-fragment: two 8-bf16 chunks at +0 and +16 (K-major column, same k order as A)
__device__ __forceinline__ void ldfragB(ABFrag& f, const u16* p) {
  f.u4[0] = *(const uint4*)(p);
  f.u4[1] = *(const uint4*)(p + 16);
}
// A-fragment: two 8-bf16 chunks at +0 and +16 (ISA 16-bit A layout)
__device__ __forceinline__ void ldfragA(ABFrag& f, const u16* p) {
  f.u4[0] = *(const uint4*)(p);
  f.u4[1] = *(const uint4*)(p + 16);
}

#define WMMA_BF16(A, B, C) \
  __builtin_amdgcn_wmma_f32_16x16x32_bf16(false, (A).v, false, (B).v, (short)0, (C), false, false)

// ---------------------------------------------------------------------------
// fp32 -> bf16 conversion (grid-stride)
// ---------------------------------------------------------------------------
__global__ void cvt_f32_bf16(const float* __restrict__ src, u16* __restrict__ dst, int n) {
  int i = blockIdx.x * blockDim.x + threadIdx.x;
  const int stride = gridDim.x * blockDim.x;
  for (; i < n; i += stride) dst[i] = f2bf(src[i]);
}

// ---------------------------------------------------------------------------
// Y(16384x512) = Xbf(16384x512) @ Wbf^T + bias.
// One wave = 32x64 output tile (2 M-subtiles x 4 N-tiles, 8 accumulators),
// K loop software-pipelined with ping-pong operand buffers so loads for
// step k+1 are in flight while WMMAs of step k execute.
// mode: 0 = bf16 -> (b,h,n,dk) head layout (Q/K)
//       1 = bf16 -> (b,h,dk,n) transposed (V)
//       2 = fp32 -> row-major d_out (final Wo projection)
// ---------------------------------------------------------------------------
__global__ void gemm512_bf16(const u16* __restrict__ Xb, const u16* __restrict__ Wb,
                             const float* __restrict__ bias, void* __restrict__ outp,
                             int mode) {
  const int lane = threadIdx.x & 31;
  const int w    = threadIdx.x >> 5;
  const int g    = blockIdx.x * 8 + w;       // 4096 wave tasks
  const int ng   = g & 7;                    // 8 groups of 64 output cols
  const int mt32 = g >> 3;                   // 512 tiles of 32 rows
  const int lh   = lane & 15;
  const int hi   = lane >> 4;

  v8f acc[2][4];
  #pragma unroll
  for (int nt = 0; nt < 4; ++nt) {
    const float bval = bias[ng * 64 + nt * 16 + lh];
    const v8f bs = (v8f){bval, bval, bval, bval, bval, bval, bval, bval};
    acc[0][nt] = bs;
    acc[1][nt] = bs;
  }

  const u16* arow0 = Xb + (size_t)(mt32 * 32 + lh) * 512 + hi * 8;
  const u16* arow1 = arow0 + 16 * 512;
  const u16* brow  = Wb + (size_t)(ng * 64 + lh) * 512 + hi * 8;

  ABFrag aP[2], bP[4], aQ[2], bQ[4];

  ldfragA(aP[0], arow0);
  ldfragA(aP[1], arow1);
  #pragma unroll
  for (int nt = 0; nt < 4; ++nt) ldfragB(bP[nt], brow + nt * 16 * 512);

  #pragma unroll
  for (int kt = 0; kt < 16; kt += 2) {
    const int kb1 = (kt + 1) * 32;
    ldfragA(aQ[0], arow0 + kb1);
    ldfragA(aQ[1], arow1 + kb1);
    #pragma unroll
    for (int nt = 0; nt < 4; ++nt) ldfragB(bQ[nt], brow + nt * 16 * 512 + kb1);

    #pragma unroll
    for (int mt = 0; mt < 2; ++mt)
      #pragma unroll
      for (int nt = 0; nt < 4; ++nt)
        acc[mt][nt] = WMMA_BF16(aP[mt], bP[nt], acc[mt][nt]);

    if (kt + 2 < 16) {
      const int kb2 = (kt + 2) * 32;
      ldfragA(aP[0], arow0 + kb2);
      ldfragA(aP[1], arow1 + kb2);
      #pragma unroll
      for (int nt = 0; nt < 4; ++nt) ldfragB(bP[nt], brow + nt * 16 * 512 + kb2);
    }

    #pragma unroll
    for (int mt = 0; mt < 2; ++mt)
      #pragma unroll
      for (int nt = 0; nt < 4; ++nt)
        acc[mt][nt] = WMMA_BF16(aQ[mt], bQ[nt], acc[mt][nt]);
  }

  // C/D layout scatter: VGPR r -> rows r (lanes 0-15) and r+8 (lanes 16-31)
  #pragma unroll
  for (int mt = 0; mt < 2; ++mt) {
    #pragma unroll
    for (int nt = 0; nt < 4; ++nt) {
      const int col = ng * 64 + nt * 16 + lh;
      #pragma unroll
      for (int r = 0; r < 8; ++r) {
        const int gm = mt32 * 32 + mt * 16 + r + hi * 8;
        const float vo = acc[mt][nt][r];
        if (mode == 2) {
          ((float*)outp)[(size_t)gm * 512 + col] = vo;
        } else {
          const int b = gm >> 12, npos = gm & 4095;
          const int h = col >> 6, d = col & 63;
          u16* o = (u16*)outp;
          if (mode == 0)
            o[(((size_t)(b * 8 + h)) * 4096 + npos) * 64 + d] = f2bf(vo);
          else
            o[(((size_t)(b * 8 + h)) * 64 + d) * 4096 + npos] = f2bf(vo);
        }
      }
    }
  }
}

// ---------------------------------------------------------------------------
// Fused flash attention. One wave = 16 queries x 2048 keys (online softmax).
// Keys are always the second half; values: section 0 -> first half, 1 -> second.
// All 16 b128 operand loads (K + V fragments) are issued at the top of each
// iteration so V streams in under the S-WMMAs and softmax VALU work.
// ---------------------------------------------------------------------------
__global__ void attn_flash(const u16* __restrict__ Qb, const u16* __restrict__ Kb,
                           const u16* __restrict__ VTb, u16* __restrict__ O1) {
  __shared__ __align__(16) u16 Ptile[8][16][32];   // 8 KB, one tile per wave

  const int lane    = threadIdx.x & 31;
  const int w       = threadIdx.x >> 5;
  const int g       = blockIdx.x * 8 + w;          // 8192 wave tasks
  const int qtile   = g & 127;
  const int section = (g >> 7) & 1;
  const int h       = (g >> 8) & 7;
  const int b       = g >> 11;
  const int lh      = lane & 15;
  const int hi      = lane >> 4;
  const float scale = 0.125f;                      // 1/sqrt(64)

  const size_t headN = (size_t)(b * 8 + h) * 4096;
  const u16* qrow = Qb + (headN + section * 2048 + qtile * 16 + lh) * 64;
  ABFrag aq0, aq1;
  ldfragA(aq0, qrow + hi * 8);
  ldfragA(aq1, qrow + 32 + hi * 8);

  const u16* krow0 = Kb + (headN + 2048 + lh) * 64 + hi * 8;
  const u16* vrow0 = VTb + ((size_t)(b * 8 + h) * 64 + lh) * 4096
                   + (section ? 2048 : 0) + hi * 8;

  float mx[8], l[8], cf[8];
  v8f O[4];
  const v8f vzero = {0.f, 0.f, 0.f, 0.f, 0.f, 0.f, 0.f, 0.f};
  #pragma unroll
  for (int r = 0; r < 8; ++r) { mx[r] = -INFINITY; l[r] = 0.f; }
  #pragma unroll
  for (int nt = 0; nt < 4; ++nt) O[nt] = vzero;

  for (int it = 0; it < 64; ++it) {                // 2048 keys, 32 per step
    const int j0 = it * 32;

    // ---- batch ALL operand loads for this step (8 K-frag + 8 V-frag b128) ----
    ABFrag bk[4], bv[4];
    const u16* kr = krow0 + (size_t)j0 * 64;
    ldfragB(bk[0], kr);                 // keys j0..j0+15,  d 0..31
    ldfragB(bk[1], kr + 32);            // keys j0..j0+15,  d 32..63
    ldfragB(bk[2], kr + 16 * 64);       // keys j0+16..+31, d 0..31
    ldfragB(bk[3], kr + 16 * 64 + 32);  // keys j0+16..+31, d 32..63
    #pragma unroll
    for (int nt = 0; nt < 4; ++nt)
      ldfragB(bv[nt], vrow0 + (size_t)nt * 16 * 4096 + j0);

    // ---- S = Q @ K^T ----
    v8f s0 = vzero, s1 = vzero;
    s0 = WMMA_BF16(aq0, bk[0], s0);
    s0 = WMMA_BF16(aq1, bk[1], s0);
    s1 = WMMA_BF16(aq0, bk[2], s1);
    s1 = WMMA_BF16(aq1, bk[3], s1);

    // ---- online softmax (rows live in 16-lane half-groups) ----
    #pragma unroll
    for (int r = 0; r < 8; ++r) {
      float rmax = fmaxf(s0[r], s1[r]);
      rmax = fmaxf(rmax, __shfl_xor(rmax, 1));
      rmax = fmaxf(rmax, __shfl_xor(rmax, 2));
      rmax = fmaxf(rmax, __shfl_xor(rmax, 4));
      rmax = fmaxf(rmax, __shfl_xor(rmax, 8));
      const float mnew = fmaxf(mx[r], rmax);
      const float c  = __expf((mx[r] - mnew) * scale);
      const float p0 = __expf((s0[r] - mnew) * scale);
      const float p1 = __expf((s1[r] - mnew) * scale);
      float rs = p0 + p1;
      rs += __shfl_xor(rs, 1);
      rs += __shfl_xor(rs, 2);
      rs += __shfl_xor(rs, 4);
      rs += __shfl_xor(rs, 8);
      l[r] = l[r] * c + rs;
      mx[r] = mnew;
      cf[r] = c;
      const int ro = r + hi * 8;
      Ptile[w][ro][lh]      = f2bf(p0);
      Ptile[w][ro][lh + 16] = f2bf(p1);
    }

    #pragma unroll
    for (int nt = 0; nt < 4; ++nt)
      #pragma unroll
      for (int r = 0; r < 8; ++r) O[nt][r] *= cf[r];

    __syncthreads();                               // uniform across all 8 waves

    ABFrag ap;                                     // P as A-fragment (16x32)
    ap.u4[0] = *(const uint4*)&Ptile[w][lh][hi * 8];
    ap.u4[1] = *(const uint4*)&Ptile[w][lh][hi * 8 + 16];

    __syncthreads();

    // ---- O += P @ V (V fragments already in flight/resident) ----
    #pragma unroll
    for (int nt = 0; nt < 4; ++nt)
      O[nt] = WMMA_BF16(ap, bv[nt], O[nt]);
  }

  // Normalize and write out1 (b, n, h*64+d) as bf16 for the Wo projection
  #pragma unroll
  for (int r = 0; r < 8; ++r) cf[r] = 1.0f / l[r];
  const size_t orow_base = ((size_t)b * 4096 + section * 2048 + qtile * 16) * 512;
  #pragma unroll
  for (int nt = 0; nt < 4; ++nt) {
    const int col = h * 64 + nt * 16 + lh;
    #pragma unroll
    for (int r = 0; r < 8; ++r) {
      const int ro = r + hi * 8;
      O1[orow_base + (size_t)ro * 512 + col] = f2bf(O[nt][r] * cf[r]);
    }
  }
}

// ---------------------------------------------------------------------------
extern "C" void kernel_launch(void* const* d_in, const int* in_sizes, int n_in,
                              void* d_out, int out_size, void* d_ws, size_t ws_size,
                              hipStream_t stream) {
  (void)in_sizes; (void)n_in; (void)out_size; (void)ws_size;
  const float* x  = (const float*)d_in[0];
  const float* Wq = (const float*)d_in[1];
  const float* bq = (const float*)d_in[2];
  const float* Wk = (const float*)d_in[3];
  const float* bk = (const float*)d_in[4];
  const float* Wv = (const float*)d_in[5];
  const float* bv = (const float*)d_in[6];
  const float* Wo = (const float*)d_in[7];
  const float* bo = (const float*)d_in[8];

  const size_t XE = (size_t)16384 * 512;   // x / q / k / v / out1 elements
  const size_t WE = (size_t)512 * 512;     // weight elements

  u16* xb  = (u16*)d_ws;
  u16* Wqb = xb  + XE;
  u16* Wkb = Wqb + WE;
  u16* Wvb = Wkb + WE;
  u16* Wob = Wvb + WE;
  u16* qb  = Wob + WE;
  u16* kbf = qb  + XE;
  u16* vTb = kbf + XE;
  u16* o1b = vTb + XE;                     // total ~86 MB of workspace

  cvt_f32_bf16<<<2048, 256, 0, stream>>>(x,  xb,  (int)XE);
  cvt_f32_bf16<<<512,  256, 0, stream>>>(Wq, Wqb, (int)WE);
  cvt_f32_bf16<<<512,  256, 0, stream>>>(Wk, Wkb, (int)WE);
  cvt_f32_bf16<<<512,  256, 0, stream>>>(Wv, Wvb, (int)WE);
  cvt_f32_bf16<<<512,  256, 0, stream>>>(Wo, Wob, (int)WE);

  gemm512_bf16<<<512, 256, 0, stream>>>(xb, Wqb, bq, qb,  0);  // Q (head layout)
  gemm512_bf16<<<512, 256, 0, stream>>>(xb, Wkb, bk, kbf, 0);  // K (head layout)
  gemm512_bf16<<<512, 256, 0, stream>>>(xb, Wvb, bv, vTb, 1);  // V (transposed)

  attn_flash<<<1024, 256, 0, stream>>>(qb, kbf, vTb, o1b);

  gemm512_bf16<<<512, 256, 0, stream>>>(o1b, Wob, bo, d_out, 2); // Wo -> fp32 out
}
